// Encoder_24532853195011
// MI455X (gfx1250) — matrix operand order repaired and checked by reading, not compiled, not run on hardware
//
#include <hip/hip_runtime.h>
#include <stdint.h>

// ---------------------------------------------------------------------------
// Types / constants
// ---------------------------------------------------------------------------
typedef __attribute__((ext_vector_type(16))) __bf16 v16bf;
typedef __attribute__((ext_vector_type(8)))  float  v8f;

constexpr int kS     = 1024;
constexpr int kDM    = 1024;
constexpr int kDEPTH = 256;
constexpr int kDFF   = 4096;
constexpr int kRows  = 8 * kS;          // 8192 rows of [*, DM]

constexpr int BM = 256, BN = 128, BK = 32;
constexpr int LDSA = BK + 2;            // 34 bf16 -> conflict-free dword reads
constexpr int LDSB = BK + 2;
constexpr int ABUF = BM * LDSA;         // elements per A buffer
constexpr int BBUF = BN * LDSB;         // elements per B buffer

// ---------------------------------------------------------------------------
// Epilogue functors
// ---------------------------------------------------------------------------
struct EpiProj {
  const float* bias;   // nullable
  const float* resid;  // nullable (fp32, same layout as output)
  float*       outF;   // nullable
  __bf16*      outB;   // nullable
  int          ldc;
  int          relu;
  long         oB, oH; // per-z output offsets: (z>>2)*oB + (z&3)*oH
  __device__ void operator()(int z, int r, int c, float v) const {
    long off = (long)(z >> 2) * oB + (long)(z & 3) * oH + (long)r * ldc + c;
    if (bias)  v += bias[c];
    if (relu)  v = v > 0.f ? v : 0.f;
    if (resid) v += resid[off];
    if (outF)  outF[off] = v;
    if (outB)  outB[off] = (__bf16)v;
  }
};

struct EpiScores {
  const float* mask;    // [B, S, S]
  float*       scores;  // [32, S, S]
  float        scale;
  __device__ void operator()(int z, int r, int c, float v) const {
    int b = z >> 2;     // z = b*H + h, H = 4
    float m = mask[(long)b * kS * kS + (long)r * kS + c];
    scores[(long)z * kS * kS + (long)r * kS + c] = v * scale - 1e9f * m;
  }
};

// ---------------------------------------------------------------------------
// Tiled bf16 WMMA GEMM: C[M x N] = A[M x K] * Bm^T, with Bm row-major [N x K]
// (all B operands are pre-transposed: weights W^T, K-matrix rows, V^T).
// Block tile 256x128, 8 waves (4x2), each wave computes 64x64 (acc[4][4]):
// LDS fragment reads drop to 4 dwords per WMMA vs 6 for a 32x64 wave tile.
// Double-buffered LDS, one barrier per K-step; global loads for tile i+1 are
// issued before the WMMA block of tile i so VMEM latency overlaps matrix math.
// grid = (N/BN, M/BM, Z). Per-z base offsets via (z>>2)*Sb + (z&3)*Sh.
// ---------------------------------------------------------------------------
template<class Epi>
__global__ __launch_bounds__(256) void gemm_kernel(
    const __bf16* __restrict__ A, int lda, long aSb, long aSh,
    const __bf16* __restrict__ Bm, int ldb, long bSb, long bSh,
    int K, Epi epi)
{
  __shared__ __bf16 sA[2 * ABUF];     // 2 x 17408 B
  __shared__ __bf16 sB[2 * BBUF];     // 2 x  8704 B

  const int tid  = threadIdx.x;
  const int lane = tid & 31;
  const int wave = tid >> 5;
  const int wM   = wave >> 1;   // 0..3  -> 64-row strip
  const int wN   = wave & 1;    // 0..1  -> 64-col strip
  const int z    = blockIdx.z;
  const int m0   = blockIdx.y * BM;
  const int n0   = blockIdx.x * BN;

  A  += (long)(z >> 2) * aSb + (long)(z & 3) * aSh;
  Bm += (long)(z >> 2) * bSb + (long)(z & 3) * bSh;

  v8f acc[4][4];
#pragma unroll
  for (int i = 0; i < 4; i++)
#pragma unroll
    for (int j = 0; j < 4; j++)
#pragma unroll
      for (int e = 0; e < 8; e++) acc[i][j][e] = 0.f;

  // A staging: 256 rows x 32 cols, 1 thread/row, full row = 4 x uint4
  // B staging: 128 rows (n) x 32 cols (k), 2 threads/row, 16 bf16 each
  const long aG  = (long)(m0 + tid) * lda;
  const int  brow = tid >> 1;
  const int  bcol = (tid & 1) * 16;
  const long bG  = (long)(n0 + brow) * ldb + bcol;

  const int hi  = lane >> 4;
  const int l15 = lane & 15;

  uint4 ra[4], rb0, rb1;

  // prologue: fetch K-tile 0 into registers
  {
    const uint4* gp = reinterpret_cast<const uint4*>(A + aG);
    ra[0] = gp[0]; ra[1] = gp[1]; ra[2] = gp[2]; ra[3] = gp[3];
    const uint4* bp = reinterpret_cast<const uint4*>(Bm + bG);
    rb0 = bp[0]; rb1 = bp[1];
  }

  int pb = 0;
  for (int k0 = 0; k0 < K; k0 += BK) {
    // ---- commit staged registers for tile k0 into LDS buffer pb ----
    {
      unsigned int* sp = reinterpret_cast<unsigned int*>(sA + pb * ABUF + tid * LDSA);
      sp[ 0] = ra[0].x; sp[ 1] = ra[0].y; sp[ 2] = ra[0].z; sp[ 3] = ra[0].w;
      sp[ 4] = ra[1].x; sp[ 5] = ra[1].y; sp[ 6] = ra[1].z; sp[ 7] = ra[1].w;
      sp[ 8] = ra[2].x; sp[ 9] = ra[2].y; sp[10] = ra[2].z; sp[11] = ra[2].w;
      sp[12] = ra[3].x; sp[13] = ra[3].y; sp[14] = ra[3].z; sp[15] = ra[3].w;
      unsigned int* tp = reinterpret_cast<unsigned int*>(sB + pb * BBUF + brow * LDSB + bcol);
      tp[0] = rb0.x; tp[1] = rb0.y; tp[2] = rb0.z; tp[3] = rb0.w;
      tp[4] = rb1.x; tp[5] = rb1.y; tp[6] = rb1.z; tp[7] = rb1.w;
    }
    __syncthreads();

    // ---- issue global loads for tile k0+BK (latency hidden by WMMA) ----
    const int kn = k0 + BK;
    if (kn < K) {
      const uint4* gp = reinterpret_cast<const uint4*>(A + aG + kn);
      ra[0] = gp[0]; ra[1] = gp[1]; ra[2] = gp[2]; ra[3] = gp[3];
      const uint4* bp = reinterpret_cast<const uint4*>(Bm + bG + kn);
      rb0 = bp[0]; rb1 = bp[1];
    }

    // ---- fragments per documented wave32 layouts ----
    // A 16x32 bf16: M=l15 (+16 per tile), K = 16*(q>>2) + 8*hi + 2*(q&3) + {0,1}
    // B 32x16 bf16: N=l15, K = 16*(q>>2) + 8*hi + 2*(q&3) + {0,1}  (pairs span K)
    union Frag { unsigned int u[8]; v16bf v; };
    Frag af[4];
    const __bf16* cA = sA + pb * ABUF;
    const __bf16* cB = sB + pb * BBUF;
    const int ar = wM * 64 + l15;
    const int bc = wN * 64 + l15;
#pragma unroll
    for (int mt = 0; mt < 4; mt++)
#pragma unroll
      for (int q = 0; q < 8; q++) {
        const int ka = ((q >> 2) << 4) + (hi << 3) + ((q & 3) << 1);
        af[mt].u[q] = *reinterpret_cast<const unsigned int*>(cA + (ar + mt * 16) * LDSA + ka);
      }
#pragma unroll
    for (int nt = 0; nt < 4; nt++) {
      Frag bf;
#pragma unroll
      for (int q = 0; q < 8; q++) {
        const int kb = ((q >> 2) << 4) + (hi << 3) + ((q & 3) << 1);
        bf.u[q] = *reinterpret_cast<const unsigned int*>(cB + (bc + nt * 16) * LDSB + kb);
      }
#pragma unroll
      for (int mt = 0; mt < 4; mt++)
        acc[mt][nt] = __builtin_amdgcn_wmma_f32_16x16x32_bf16(
            false, af[mt].v, false, bf.v, (short)0, acc[mt][nt], false, false);
    }

    pb ^= 1;
  }

  // C/D layout: VGPR i -> M = 8*hi + i (within 16-tile), N = l15
#pragma unroll
  for (int mt = 0; mt < 4; mt++)
#pragma unroll
    for (int nt = 0; nt < 4; nt++)
#pragma unroll
      for (int i = 0; i < 8; i++) {
        int r = m0 + wM * 64 + mt * 16 + hi * 8 + i;
        int c = n0 + wN * 64 + nt * 16 + l15;
        epi(z, r, c, acc[mt][nt][i]);
      }
}

// ---------------------------------------------------------------------------
// Transposing converters (one-time, bandwidth-trivial)
// ---------------------------------------------------------------------------
// W[K x N] fp32 -> W^T[N x K] bf16, 32x32 LDS tiles. grid = (N/32, K/32)
__global__ __launch_bounds__(256) void wtrans_kernel(const float* __restrict__ w,
                                                     __bf16* __restrict__ wt,
                                                     int K, int N)
{
  __shared__ float t[32][33];
  const int k0 = blockIdx.y * 32;
  const int n0 = blockIdx.x * 32;
  const int r  = threadIdx.x >> 3;
  const int c4 = (threadIdx.x & 7) * 4;
  float4 d = *reinterpret_cast<const float4*>(w + (long)(k0 + r) * N + n0 + c4);
  t[r][c4 + 0] = d.x; t[r][c4 + 1] = d.y; t[r][c4 + 2] = d.z; t[r][c4 + 3] = d.w;
  __syncthreads();
  union { uint2 u; __bf16 h4[4]; } o;
  o.h4[0] = (__bf16)t[c4 + 0][r]; o.h4[1] = (__bf16)t[c4 + 1][r];
  o.h4[2] = (__bf16)t[c4 + 2][r]; o.h4[3] = (__bf16)t[c4 + 3][r];
  *reinterpret_cast<uint2*>(wt + (long)(n0 + r) * K + k0 + c4) = o.u;
}

// v[B,S,DM] bf16 (head h cols) -> vT[b*4+h][DEPTH][S] bf16. grid = (8, 32, 32)
__global__ __launch_bounds__(256) void vtrans_kernel(const __bf16* __restrict__ v,
                                                     __bf16* __restrict__ vt)
{
  __shared__ __bf16 t[32][34];
  const int z  = blockIdx.z;          // b*4 + h
  const int d0 = blockIdx.x * 32;
  const int s0 = blockIdx.y * 32;
  const int b  = z >> 2, h = z & 3;
  const int r  = threadIdx.x >> 3;
  const int c4 = (threadIdx.x & 7) * 4;
  const __bf16* src = v + (long)b * kS * kDM + (long)(s0 + r) * kDM + h * kDEPTH + d0 + c4;
  union { uint2 u; __bf16 h4[4]; } in;
  in.u = *reinterpret_cast<const uint2*>(src);
  t[r][c4 + 0] = in.h4[0]; t[r][c4 + 1] = in.h4[1];
  t[r][c4 + 2] = in.h4[2]; t[r][c4 + 3] = in.h4[3];
  __syncthreads();
  union { uint2 u; __bf16 h4[4]; } o;
  o.h4[0] = t[c4 + 0][r]; o.h4[1] = t[c4 + 1][r];
  o.h4[2] = t[c4 + 2][r]; o.h4[3] = t[c4 + 3][r];
  *reinterpret_cast<uint2*>(vt + (long)z * kDEPTH * kS + (long)(d0 + r) * kS + s0 + c4) = o.u;
}

// ---------------------------------------------------------------------------
// Elementwise / reduction kernels (bandwidth-bound; vectorized)
// ---------------------------------------------------------------------------
// 8 fp32 -> 8 bf16 per thread
__global__ __launch_bounds__(256) void cvt8_kernel(const float* __restrict__ src,
                                                   __bf16* __restrict__ dst, long n8)
{
  long i = (long)blockIdx.x * 256 + threadIdx.x;
  if (i >= n8) return;
  const float4* s4 = reinterpret_cast<const float4*>(src) + i * 2;
  float4 x0 = s4[0], x1 = s4[1];
  union { uint4 u; __bf16 h[8]; } o;
  o.h[0] = (__bf16)x0.x; o.h[1] = (__bf16)x0.y; o.h[2] = (__bf16)x0.z; o.h[3] = (__bf16)x0.w;
  o.h[4] = (__bf16)x1.x; o.h[5] = (__bf16)x1.y; o.h[6] = (__bf16)x1.z; o.h[7] = (__bf16)x1.w;
  reinterpret_cast<uint4*>(dst)[i] = o.u;
}

// one wave per row of 1024; 8 rows per block; float4 loads, packed bf16 stores
__global__ __launch_bounds__(256) void softmax_kernel(const float* __restrict__ S,
                                                      __bf16* __restrict__ P)
{
  const int lane = threadIdx.x & 31;
  const int w    = threadIdx.x >> 5;
  const long row = (long)blockIdx.x * 8 + w;
  const float4* s4 = reinterpret_cast<const float4*>(S + row * kS);
  float v[32];
  float mx = -3.4e38f;
#pragma unroll
  for (int i = 0; i < 8; i++) {
    float4 t = s4[lane + i * 32];
    v[4*i+0] = t.x; v[4*i+1] = t.y; v[4*i+2] = t.z; v[4*i+3] = t.w;
    mx = fmaxf(mx, fmaxf(fmaxf(t.x, t.y), fmaxf(t.z, t.w)));
  }
#pragma unroll
  for (int off = 16; off > 0; off >>= 1) mx = fmaxf(mx, __shfl_xor(mx, off, 32));
  float sum = 0.f;
#pragma unroll
  for (int i = 0; i < 32; i++) { v[i] = __expf(v[i] - mx); sum += v[i]; }
#pragma unroll
  for (int off = 16; off > 0; off >>= 1) sum += __shfl_xor(sum, off, 32);
  const float inv = 1.f / sum;
  uint2* p2 = reinterpret_cast<uint2*>(P + row * kS);
#pragma unroll
  for (int i = 0; i < 8; i++) {
    union { uint2 u; __bf16 h[4]; } o;
    o.h[0] = (__bf16)(v[4*i+0] * inv); o.h[1] = (__bf16)(v[4*i+1] * inv);
    o.h[2] = (__bf16)(v[4*i+2] * inv); o.h[3] = (__bf16)(v[4*i+3] * inv);
    p2[lane + i * 32] = o.u;
  }
}

__global__ void zero_kernel(float* p, int n)
{
  int i = blockIdx.x * 256 + threadIdx.x;
  if (i < n) p[i] = 0.f;
}

// 256 blocks x 32 rows; each thread owns 4 consecutive channels (float4)
__global__ __launch_bounds__(256) void bn_partial_kernel(const float* __restrict__ y,
                                                         float* __restrict__ sums)
{
  float4 s1 = {0.f, 0.f, 0.f, 0.f};
  float4 s2 = {0.f, 0.f, 0.f, 0.f};
  const int r0 = blockIdx.x * 32;
  const float4* y4 = reinterpret_cast<const float4*>(y);
  for (int r = 0; r < 32; r++) {
    float4 v = y4[(long)(r0 + r) * (kDM / 4) + threadIdx.x];
    s1.x += v.x; s1.y += v.y; s1.z += v.z; s1.w += v.w;
    s2.x += v.x * v.x; s2.y += v.y * v.y; s2.z += v.z * v.z; s2.w += v.w * v.w;
  }
  const int c = threadIdx.x * 4;
  atomicAdd(&sums[c + 0], s1.x); atomicAdd(&sums[c + 1], s1.y);
  atomicAdd(&sums[c + 2], s1.z); atomicAdd(&sums[c + 3], s1.w);
  atomicAdd(&sums[kDM + c + 0], s2.x); atomicAdd(&sums[kDM + c + 1], s2.y);
  atomicAdd(&sums[kDM + c + 2], s2.z); atomicAdd(&sums[kDM + c + 3], s2.w);
}

__global__ void bn_final_kernel(const float* __restrict__ sums, float* __restrict__ mi)
{
  int c = blockIdx.x * 256 + threadIdx.x;
  if (c >= kDM) return;
  float m   = sums[c] * (1.f / 8192.f);
  float var = sums[kDM + c] * (1.f / 8192.f) - m * m;  // biased
  mi[c]       = m;
  mi[kDM + c] = rsqrtf(var + 1e-5f);
}

// float4 per thread: 4 consecutive channels
__global__ __launch_bounds__(256) void bn_apply_kernel(const float* __restrict__ y,
                                                       const float* __restrict__ mi,
                                                       const float* __restrict__ g,
                                                       const float* __restrict__ beta,
                                                       float* outF, __bf16* outB)
{
  long i = (long)blockIdx.x * 256 + threadIdx.x;   // float4 index
  int  c = (int)((i * 4) & (kDM - 1));
  float4 v = reinterpret_cast<const float4*>(y)[i];
  float4 r;
  r.x = (v.x - mi[c + 0]) * mi[kDM + c + 0] * g[c + 0] + beta[c + 0];
  r.y = (v.y - mi[c + 1]) * mi[kDM + c + 1] * g[c + 1] + beta[c + 1];
  r.z = (v.z - mi[c + 2]) * mi[kDM + c + 2] * g[c + 2] + beta[c + 2];
  r.w = (v.w - mi[c + 3]) * mi[kDM + c + 3] * g[c + 3] + beta[c + 3];
  if (outF) reinterpret_cast<float4*>(outF)[i] = r;
  if (outB) {
    union { uint2 u; __bf16 h[4]; } o;
    o.h[0] = (__bf16)r.x; o.h[1] = (__bf16)r.y; o.h[2] = (__bf16)r.z; o.h[3] = (__bf16)r.w;
    reinterpret_cast<uint2*>(outB)[i] = o.u;
  }
}

// ---------------------------------------------------------------------------
// Host launcher
// ---------------------------------------------------------------------------
extern "C" void kernel_launch(void* const* d_in, const int* in_sizes, int n_in,
                              void* d_out, int out_size, void* d_ws, size_t ws_size,
                              hipStream_t stream)
{
  (void)in_sizes; (void)n_in; (void)out_size; (void)ws_size;

  const float* x    = (const float*)d_in[0];
  const float* mask = (const float*)d_in[1];
  const float* wq = (const float*)d_in[2];  const float* bq = (const float*)d_in[3];
  const float* wk = (const float*)d_in[4];  const float* bk = (const float*)d_in[5];
  const float* wv = (const float*)d_in[6];  const float* bv = (const float*)d_in[7];
  const float* wo = (const float*)d_in[8];  const float* bo = (const float*)d_in[9];
  const float* w1 = (const float*)d_in[10]; const float* b1 = (const float*)d_in[11];
  const float* w2 = (const float*)d_in[12]; const float* b2 = (const float*)d_in[13];
  const float* g1 = (const float*)d_in[14]; const float* be1 = (const float*)d_in[15];
  const float* g2 = (const float*)d_in[16]; const float* be2 = (const float*)d_in[17];
  float* out = (float*)d_out;

  // Workspace layout (peak ~296 MB; overlapping regions have disjoint lifetimes)
  char* ws = (char*)d_ws;
  const size_t MB = 1ull << 20;
  __bf16* wqT = (__bf16*)(ws + 0 * MB);     // 2 MB  [N=1024][K=1024]
  __bf16* wkT = (__bf16*)(ws + 2 * MB);
  __bf16* wvT = (__bf16*)(ws + 4 * MB);
  __bf16* woT = (__bf16*)(ws + 6 * MB);
  __bf16* w1T = (__bf16*)(ws + 8 * MB);     // 8 MB  [4096][1024]
  __bf16* w2T = (__bf16*)(ws + 16 * MB);    // 8 MB  [1024][4096]
  __bf16* xb  = (__bf16*)(ws + 24 * MB);    // 16 MB, dead after QKV
  __bf16* ob  = (__bf16*)(ws + 24 * MB);    // reuse xb region
  __bf16* qb  = (__bf16*)(ws + 40 * MB);    // 16 MB, dead after scores
  __bf16* kbm = (__bf16*)(ws + 56 * MB);    // 16 MB, dead after scores
  __bf16* vb  = (__bf16*)(ws + 72 * MB);    // 16 MB, dead after vtrans
  __bf16* hb  = (__bf16*)(ws + 40 * MB);    // 64 MB, reuse q/k/v + dead scores head
  float*  sc  = (float*)(ws + 88 * MB);     // 128 MB scores, dead after softmax
  float*  y1  = (float*)(ws + 104 * MB);    // 32 MB (in dead scores region)
  float*  o1f = (float*)(ws + 136 * MB);    // 32 MB
  __bf16* o1b = (__bf16*)(ws + 168 * MB);   // 16 MB
  float*  y2  = (float*)(ws + 184 * MB);    // 32 MB
  __bf16* Pb  = (__bf16*)(ws + 216 * MB);   // 64 MB, dead after attn-out GEMM
  __bf16* vT  = (__bf16*)(ws + 280 * MB);   // 16 MB [b*4+h][DEPTH][S]
  float*  sums = (float*)(ws + 296 * MB);   // 8 KB
  float*  mi   = (float*)(ws + 296 * MB + 8192);

  const long SDM = (long)kS * kDM;          // per-batch stride of [S, DM]
  const long SS  = (long)kS * kS;
  const long DS  = (long)kDEPTH * kS;       // per-head stride of vT

  // --- convert / transpose inputs to bf16 ---
  cvt8_kernel<<<4096, 256, 0, stream>>>(x, xb, (long)kRows * kDM / 8);
  wtrans_kernel<<<dim3(32, 32), 256, 0, stream>>>(wq, wqT, kDM, kDM);
  wtrans_kernel<<<dim3(32, 32), 256, 0, stream>>>(wk, wkT, kDM, kDM);
  wtrans_kernel<<<dim3(32, 32), 256, 0, stream>>>(wv, wvT, kDM, kDM);
  wtrans_kernel<<<dim3(32, 32), 256, 0, stream>>>(wo, woT, kDM, kDM);
  wtrans_kernel<<<dim3(128, 32), 256, 0, stream>>>(w1, w1T, kDM, kDFF);   // -> [4096][1024]
  wtrans_kernel<<<dim3(32, 128), 256, 0, stream>>>(w2, w2T, kDFF, kDM);   // -> [1024][4096]

  // --- Q, K, V projections (bf16 out) ---
  {
    EpiProj e{}; e.ldc = kDM;
    e.bias = bq; e.outB = qb;
    gemm_kernel<EpiProj><<<dim3(kDM / BN, kRows / BM, 1), 256, 0, stream>>>(
        xb, kDM, 0, 0, wqT, kDM, 0, 0, kDM, e);
    e.bias = bk; e.outB = kbm;
    gemm_kernel<EpiProj><<<dim3(kDM / BN, kRows / BM, 1), 256, 0, stream>>>(
        xb, kDM, 0, 0, wkT, kDM, 0, 0, kDM, e);
    e.bias = bv; e.outB = vb;
    gemm_kernel<EpiProj><<<dim3(kDM / BN, kRows / BM, 1), 256, 0, stream>>>(
        xb, kDM, 0, 0, wvT, kDM, 0, 0, kDM, e);
  }

  // --- V -> V^T per head (so P*V stages N-major too) ---
  vtrans_kernel<<<dim3(8, 32, 32), 256, 0, stream>>>(vb, vT);

  // --- scores = Q K^T * scale + mask * (-1e9), per (b,h), fp32 ---
  {
    EpiScores e{mask, sc, 0.0625f};  // 1/sqrt(256)
    gemm_kernel<EpiScores><<<dim3(kS / BN, kS / BM, 32), 256, 0, stream>>>(
        qb, kDM, SDM, kDEPTH, kbm, kDM, SDM, kDEPTH, kDEPTH, e);
  }

  // --- softmax rows -> P (bf16) ---
  softmax_kernel<<<4096, 256, 0, stream>>>(sc, Pb);

  // --- O = P V per (b,h), written into merged [B,S,DM] layout (bf16) ---
  {
    EpiProj e{}; e.ldc = kDM; e.outB = ob; e.oB = SDM; e.oH = kDEPTH;
    gemm_kernel<EpiProj><<<dim3(kDEPTH / BN, kS / BM, 32), 256, 0, stream>>>(
        Pb, kS, 4 * SS, SS, vT, kS, 4 * DS, DS, kS, e);
  }

  // --- O projection + residual x -> y1 (fp32) ---
  {
    EpiProj e{}; e.ldc = kDM; e.bias = bo; e.resid = x; e.outF = y1;
    gemm_kernel<EpiProj><<<dim3(kDM / BN, kRows / BM, 1), 256, 0, stream>>>(
        ob, kDM, 0, 0, woT, kDM, 0, 0, kDM, e);
  }

  // --- BatchNorm 1 ---
  zero_kernel<<<8, 256, 0, stream>>>(sums, 2 * kDM);
  bn_partial_kernel<<<256, 256, 0, stream>>>(y1, sums);
  bn_final_kernel<<<4, 256, 0, stream>>>(sums, mi);
  bn_apply_kernel<<<8192, 256, 0, stream>>>(y1, mi, g1, be1, o1f, o1b);

  // --- FFN1: relu(out1 @ w1 + b1) -> hb (bf16) ---
  {
    EpiProj e{}; e.ldc = kDFF; e.bias = b1; e.relu = 1; e.outB = hb;
    gemm_kernel<EpiProj><<<dim3(kDFF / BN, kRows / BM, 1), 256, 0, stream>>>(
        o1b, kDM, 0, 0, w1T, kDM, 0, 0, kDM, e);
  }

  // --- FFN2: hb @ w2 + b2 + out1 -> y2 (fp32) ---
  {
    EpiProj e{}; e.ldc = kDM; e.bias = b2; e.resid = o1f; e.outF = y2;
    gemm_kernel<EpiProj><<<dim3(kDM / BN, kRows / BM, 1), 256, 0, stream>>>(
        hb, kDFF, 0, 0, w2T, kDFF, 0, 0, kDFF, e);
  }

  // --- BatchNorm 2 -> final output ---
  zero_kernel<<<8, 256, 0, stream>>>(sums, 2 * kDM);
  bn_partial_kernel<<<256, 256, 0, stream>>>(y2, sums);
  bn_final_kernel<<<4, 256, 0, stream>>>(sums, mi);
  bn_apply_kernel<<<8192, 256, 0, stream>>>(y2, mi, g2, be2, out, nullptr);
}
